// MultiHeadAttention_17987323036524
// MI455X (gfx1250) — matrix operand order repaired and checked by reading, not compiled, not run on hardware
//
#include <hip/hip_runtime.h>
#include <hip/hip_bf16.h>

typedef __attribute__((ext_vector_type(16))) _Float16 v16h;
typedef __attribute__((ext_vector_type(8)))  _Float16 v8h;
typedef __attribute__((ext_vector_type(8)))  float    v8f;

#define EMBED    1024
#define HEADS    16
#define HEAD_DIM 64
#define BATCH    2
#define SEQ      2048

// ---------------------------------------------------------------------------
// WMMA helpers (CDNA5 wave32, V_WMMA_F32_16X16X32_F16)
// ---------------------------------------------------------------------------
__device__ __forceinline__ v8f wmma16(v16h a, v16h b, v8f c) {
  return __builtin_amdgcn_wmma_f32_16x16x32_f16(
      /*neg_a=*/false, a, /*neg_b=*/false, b,
      /*c_mod=*/(short)0, c, /*reuse_a=*/false, /*reuse_b=*/false);
}

// A fragment, 16x32 f16.  ISA layout: lane L holds row M=L&15.
// lanes 0-15: VGPR0-3 = K 0..7,  VGPR4-7 = K 16..23
// lanes16-31: VGPR0-3 = K 8..15, VGPR4-7 = K 24..31
// -> two contiguous 16B LDS reads per lane.
__device__ __forceinline__ v16h frag_a(const _Float16* base, int ld, int lane) {
  const int row = lane & 15, half = lane >> 4;
  const _Float16* p = base + row * ld + half * 8;
  union { v16h v; v8h h[2]; } u;
  u.h[0] = *(const v8h*)(p);        // K = half*8 .. +7
  u.h[1] = *(const v8h*)(p + 16);   // K = 16 + half*8 .. +7
  return u.v;
}

// B fragment, 32x16 f16, sourced from an LDS tile stored TRANSPOSED: Bt[n][k].
// lane L holds column N=L&15; K = (L>>4)*8 + t (t<8), 16 + (L>>4)*8 + t-8 (t>=8)
// -> two runs of 8 contiguous halfs = two 16B LDS reads per lane.
__device__ __forceinline__ v16h frag_b(const _Float16* base, int ld, int lane) {
  const int n = lane & 15, half = lane >> 4;
  const _Float16* p = base + n * ld + half * 8;
  union { v16h v; v8h h[2]; } u;
  u.h[0] = *(const v8h*)(p);
  u.h[1] = *(const v8h*)(p + 16);
  return u.v;
}

// ---------------------------------------------------------------------------
// Kernel 1: fused QKV projection.  C[4096,3072] = X[4096,1024] * Wqkv^T + b
// Output scattered to Q/K/V f16 workspaces in (B,H,S,hd); 1/sqrt(hd) folded
// into Q.
// ---------------------------------------------------------------------------
__global__ __launch_bounds__(256) void qkv_gemm_kernel(
    const float* __restrict__ X, const float* __restrict__ W,
    const float* __restrict__ bias,
    _Float16* __restrict__ Qw, _Float16* __restrict__ Kw,
    _Float16* __restrict__ Vw) {
  __shared__ alignas(16) _Float16 As[128][40];  // [m][k], padded
  __shared__ alignas(16) _Float16 Bs[128][40];  // [n][k] == W row-major slice

  const int tid = threadIdx.x;
  const int lane = tid & 31, wave = tid >> 5;
  const int wm = wave >> 1, wn = wave & 1;      // 4x2 wave grid
  const int m0 = blockIdx.y * 128, n0 = blockIdx.x * 128;

  v8f acc[2][4];
#pragma unroll
  for (int i = 0; i < 2; ++i)
#pragma unroll
    for (int j = 0; j < 4; ++j) { v8f z = {}; acc[i][j] = z; }

  const int lrow = tid >> 1;
  const int lcol = (tid & 1) * 16;

  for (int k0 = 0; k0 < EMBED; k0 += 32) {
    __syncthreads();
    const float* xa = X + (size_t)(m0 + lrow) * EMBED + k0 + lcol;
    const float* wb = W + (size_t)(n0 + lrow) * EMBED + k0 + lcol;
#pragma unroll
    for (int u = 0; u < 16; u += 4) {
      float4 xv = *(const float4*)(xa + u);
      float4 wv = *(const float4*)(wb + u);
      As[lrow][lcol + u + 0] = (_Float16)xv.x;
      As[lrow][lcol + u + 1] = (_Float16)xv.y;
      As[lrow][lcol + u + 2] = (_Float16)xv.z;
      As[lrow][lcol + u + 3] = (_Float16)xv.w;
      Bs[lrow][lcol + u + 0] = (_Float16)wv.x;
      Bs[lrow][lcol + u + 1] = (_Float16)wv.y;
      Bs[lrow][lcol + u + 2] = (_Float16)wv.z;
      Bs[lrow][lcol + u + 3] = (_Float16)wv.w;
    }
    __syncthreads();

    v16h af[2], bf[4];
#pragma unroll
    for (int i = 0; i < 2; ++i) af[i] = frag_a(&As[wm * 32 + i * 16][0], 40, lane);
#pragma unroll
    for (int j = 0; j < 4; ++j) bf[j] = frag_b(&Bs[wn * 64 + j * 16][0], 40, lane);
#pragma unroll
    for (int i = 0; i < 2; ++i)
#pragma unroll
      for (int j = 0; j < 4; ++j) acc[i][j] = wmma16(af[i], bf[j], acc[i][j]);
  }

  // Epilogue: C layout -> lane holds (M = r + 8*(lane>>4), N = lane&15)
  const int nl = lane & 15, hl = lane >> 4;
#pragma unroll
  for (int j = 0; j < 4; ++j) {
    const int n = n0 + wn * 64 + j * 16 + nl;
    const float bv = bias[n];
    const int part = n >> 10;
    const int rem = n & 1023;
    const int h = rem >> 6, d = rem & 63;
    _Float16* dst = (part == 0) ? Qw : (part == 1) ? Kw : Vw;
    const float sc = (part == 0) ? 0.125f : 1.0f;  // 1/sqrt(64) folded into Q
#pragma unroll
    for (int i = 0; i < 2; ++i) {
#pragma unroll
      for (int r = 0; r < 8; ++r) {
        const int m = m0 + wm * 32 + i * 16 + hl * 8 + r;
        const int b = m >> 11, s = m & 2047;
        const float val = (acc[i][j][r] + bv) * sc;
        dst[((size_t)((b * HEADS + h) * SEQ + s) << 6) + d] = (_Float16)val;
      }
    }
  }
}

// ---------------------------------------------------------------------------
// Kernel 2: flash attention.  One (b,h) x 128-query block per workgroup.
// 8 waves, each owns 16 query rows; online softmax in registers.
// ---------------------------------------------------------------------------
__global__ __launch_bounds__(256) void attn_kernel(
    const _Float16* __restrict__ Qw, const _Float16* __restrict__ Kw,
    const _Float16* __restrict__ Vw, _Float16* __restrict__ Aout) {
  __shared__ alignas(16) _Float16 Qs[128][72];       // [q][hd]
  __shared__ alignas(16) _Float16 Ks[64][72];        // [key][hd]  (B^T for QK^T)
  __shared__ alignas(16) _Float16 Vt[64][72];        // [hd][key]  (B^T for P*V)
  __shared__ alignas(16) _Float16 Ps[8][16][72];     // per-wave P relayout

  const int tid = threadIdx.x;
  const int lane = tid & 31, wave = tid >> 5;
  const int bh = blockIdx.y;
  const _Float16* Qp = Qw + ((size_t)bh * SEQ + (size_t)blockIdx.x * 128) * HEAD_DIM;
  const _Float16* Kp = Kw + (size_t)bh * SEQ * HEAD_DIM;
  const _Float16* Vp = Vw + (size_t)bh * SEQ * HEAD_DIM;

  {  // stage Q block (128x64 f16)
    const int row = tid >> 1, col = (tid & 1) * 32;
#pragma unroll
    for (int u = 0; u < 32; u += 8)
      *(v8h*)&Qs[row][col + u] = *(const v8h*)(Qp + row * HEAD_DIM + col + u);
  }
  __syncthreads();

  // Q fragments are loop-invariant
  v16h qa0 = frag_a(&Qs[wave * 16][0], 72, lane);
  v16h qa1 = frag_a(&Qs[wave * 16][32], 72, lane);

  float mrow[8], lrowv[8];
  v8f oacc[4];
#pragma unroll
  for (int r = 0; r < 8; ++r) { mrow[r] = -1e30f; lrowv[r] = 0.f; }
#pragma unroll
  for (int nt = 0; nt < 4; ++nt) { v8f z = {}; oacc[nt] = z; }

  const int nl = lane & 15, hl = lane >> 4;
  const int krow = tid >> 2, kcol = (tid & 3) * 16;

  for (int j0 = 0; j0 < SEQ; j0 += 64) {
    __syncthreads();  // protect Ks/Vt against prior-iteration readers
#pragma unroll
    for (int u = 0; u < 16; u += 8)
      *(v8h*)&Ks[krow][kcol + u] =
          *(const v8h*)(Kp + (size_t)(j0 + krow) * HEAD_DIM + kcol + u);
#pragma unroll
    for (int u = 0; u < 16; ++u)  // V transposed: Vt[hd][key]
      Vt[kcol + u][krow] = Vp[(size_t)(j0 + krow) * HEAD_DIM + kcol + u];
    __syncthreads();

    // scores: 16 q x 64 keys (4 N-tiles, 2 chained WMMAs each over hd=64)
    v8f sc[4];
#pragma unroll
    for (int nt = 0; nt < 4; ++nt) {
      v8f s = {};
      v16h kb0 = frag_b(&Ks[nt * 16][0], 72, lane);
      v16h kb1 = frag_b(&Ks[nt * 16][32], 72, lane);
      s = wmma16(qa0, kb0, s);
      s = wmma16(qa1, kb1, s);
      sc[nt] = s;
    }

    // online softmax; row (r, lane-half) values live on 16 lanes of the half
    float alpha[8];
#pragma unroll
    for (int r = 0; r < 8; ++r) {
      float mx = fmaxf(fmaxf(sc[0][r], sc[1][r]), fmaxf(sc[2][r], sc[3][r]));
      mx = fmaxf(mx, __shfl_xor(mx, 1, 32));
      mx = fmaxf(mx, __shfl_xor(mx, 2, 32));
      mx = fmaxf(mx, __shfl_xor(mx, 4, 32));
      mx = fmaxf(mx, __shfl_xor(mx, 8, 32));
      const float nm = fmaxf(mrow[r], mx);
      alpha[r] = __expf(mrow[r] - nm);
      mrow[r] = nm;
      float psum = 0.f;
#pragma unroll
      for (int nt = 0; nt < 4; ++nt) {
        const float e = __expf(sc[nt][r] - nm);
        sc[nt][r] = e;
        psum += e;
      }
      psum += __shfl_xor(psum, 1, 32);
      psum += __shfl_xor(psum, 2, 32);
      psum += __shfl_xor(psum, 4, 32);
      psum += __shfl_xor(psum, 8, 32);
      lrowv[r] = lrowv[r] * alpha[r] + psum;
    }

    // re-layout P (C-fragment) into A-fragment via per-wave LDS; rescale O
#pragma unroll
    for (int nt = 0; nt < 4; ++nt)
#pragma unroll
      for (int r = 0; r < 8; ++r) {
        Ps[wave][hl * 8 + r][nt * 16 + nl] = (_Float16)sc[nt][r];
        oacc[nt][r] *= alpha[r];
      }
    __syncthreads();

    v16h pa0 = frag_a(&Ps[wave][0][0], 72, lane);
    v16h pa1 = frag_a(&Ps[wave][0][32], 72, lane);
#pragma unroll
    for (int nt = 0; nt < 4; ++nt) {
      v16h vb0 = frag_b(&Vt[nt * 16][0], 72, lane);
      v16h vb1 = frag_b(&Vt[nt * 16][32], 72, lane);
      oacc[nt] = wmma16(pa0, vb0, oacc[nt]);
      oacc[nt] = wmma16(pa1, vb1, oacc[nt]);
    }
  }

  // epilogue: O /= l ; write f16 attn rows in (B,S,E) for the output GEMM
  const int b = bh >> 4, h = bh & 15;
#pragma unroll
  for (int nt = 0; nt < 4; ++nt)
#pragma unroll
    for (int r = 0; r < 8; ++r) {
      const int qrow = blockIdx.x * 128 + wave * 16 + hl * 8 + r;
      const int col = h * HEAD_DIM + nt * 16 + nl;
      Aout[((size_t)(b * SEQ + qrow) << 10) + col] =
          (_Float16)(oacc[nt][r] / lrowv[r]);
    }
}

// ---------------------------------------------------------------------------
// Kernel 3: output projection. out[4096,1024] = attn(f16) * Wout^T + b (fp32)
// ---------------------------------------------------------------------------
__global__ __launch_bounds__(256) void out_gemm_kernel(
    const _Float16* __restrict__ A, const float* __restrict__ W,
    const float* __restrict__ bias, float* __restrict__ out) {
  __shared__ alignas(16) _Float16 As[128][40];
  __shared__ alignas(16) _Float16 Bs[128][40];

  const int tid = threadIdx.x;
  const int lane = tid & 31, wave = tid >> 5;
  const int wm = wave >> 1, wn = wave & 1;
  const int m0 = blockIdx.y * 128, n0 = blockIdx.x * 128;

  v8f acc[2][4];
#pragma unroll
  for (int i = 0; i < 2; ++i)
#pragma unroll
    for (int j = 0; j < 4; ++j) { v8f z = {}; acc[i][j] = z; }

  const int lrow = tid >> 1;
  const int lcol = (tid & 1) * 16;

  for (int k0 = 0; k0 < EMBED; k0 += 32) {
    __syncthreads();
    const _Float16* aa = A + (size_t)(m0 + lrow) * EMBED + k0 + lcol;
    *(v8h*)&As[lrow][lcol + 0] = *(const v8h*)(aa + 0);
    *(v8h*)&As[lrow][lcol + 8] = *(const v8h*)(aa + 8);
    const float* wb = W + (size_t)(n0 + lrow) * EMBED + k0 + lcol;
#pragma unroll
    for (int u = 0; u < 16; u += 4) {
      float4 wv = *(const float4*)(wb + u);
      Bs[lrow][lcol + u + 0] = (_Float16)wv.x;
      Bs[lrow][lcol + u + 1] = (_Float16)wv.y;
      Bs[lrow][lcol + u + 2] = (_Float16)wv.z;
      Bs[lrow][lcol + u + 3] = (_Float16)wv.w;
    }
    __syncthreads();

    v16h af[2], bf[4];
#pragma unroll
    for (int i = 0; i < 2; ++i) af[i] = frag_a(&As[wm * 32 + i * 16][0], 40, lane);
#pragma unroll
    for (int j = 0; j < 4; ++j) bf[j] = frag_b(&Bs[wn * 64 + j * 16][0], 40, lane);
#pragma unroll
    for (int i = 0; i < 2; ++i)
#pragma unroll
      for (int j = 0; j < 4; ++j) acc[i][j] = wmma16(af[i], bf[j], acc[i][j]);
  }

  const int nl = lane & 15, hl = lane >> 4;
#pragma unroll
  for (int j = 0; j < 4; ++j) {
    const int n = n0 + wn * 64 + j * 16 + nl;
    const float bv = bias[n];
#pragma unroll
    for (int i = 0; i < 2; ++i)
#pragma unroll
      for (int r = 0; r < 8; ++r) {
        const int m = m0 + wm * 32 + i * 16 + hl * 8 + r;
        out[(size_t)m * EMBED + n] = acc[i][j][r] + bv;
      }
  }
}

// ---------------------------------------------------------------------------
extern "C" void kernel_launch(void* const* d_in, const int* in_sizes, int n_in,
                              void* d_out, int out_size, void* d_ws,
                              size_t ws_size, hipStream_t stream) {
  (void)in_sizes; (void)n_in; (void)out_size; (void)ws_size;
  const float* X    = (const float*)d_in[0];  // (2,2048,1024)
  const float* Wqkv = (const float*)d_in[1];  // (3072,1024)
  const float* bqkv = (const float*)d_in[2];  // (3072,)
  const float* Wout = (const float*)d_in[3];  // (1024,1024)
  const float* bout = (const float*)d_in[4];  // (1024,)
  float* out = (float*)d_out;

  const size_t PER = (size_t)BATCH * HEADS * SEQ * HEAD_DIM;  // 4M f16 elems
  _Float16* ws = (_Float16*)d_ws;
  _Float16* Qw = ws;
  _Float16* Kw = ws + PER;
  _Float16* Vw = ws + 2 * PER;
  _Float16* Aw = ws + 3 * PER;  // (B*S, E) f16 attention output

  qkv_gemm_kernel<<<dim3(3 * EMBED / 128, BATCH * SEQ / 128), 256, 0, stream>>>(
      X, Wqkv, bqkv, Qw, Kw, Vw);
  attn_kernel<<<dim3(SEQ / 128, BATCH * HEADS), 256, 0, stream>>>(Qw, Kw, Vw, Aw);
  out_gemm_kernel<<<dim3(EMBED / 128, BATCH * SEQ / 128), 256, 0, stream>>>(
      Aw, Wout, bout, out);
}
